// Fconv_PCA_10857677324903
// MI455X (gfx1250) — matrix operand order repaired and checked by reading, not compiled, not run on hardware
//
#include <hip/hip_runtime.h>
#include <hip/hip_bf16.h>

typedef __attribute__((ext_vector_type(16))) _Float16 v16h;
typedef __attribute__((ext_vector_type(8)))  _Float16 v8h;
typedef __attribute__((ext_vector_type(8)))  float    v8f;

#define BATCH   8
#define CIN     64
#define COUT    64
#define HH      192
#define WW      192
#define SZP     5
#define KPIX    25
#define TRAN    8
#define LDSPITCH 72           // padded ci pitch (halves) to spread LDS banks
#define LDSCOLS 68            // 64-pixel tile + 2*2 halo

// ---------------------------------------------------------------------------
// Stage 1: steerable PCA basis, basis[b][i][j][o][k], 8*5*5*8*25 = 40000 f32
// ---------------------------------------------------------------------------
__device__ __forceinline__ float bicubic_f(float v) {
    float a  = fabsf(v);
    float a2 = a * a, a3 = a2 * a;
    float t1 =  1.5f * a3 - 2.5f * a2 + 1.0f;
    float t2 = -0.5f * a3 + 2.5f * a2 - 4.0f * a + 2.0f;
    float r = 0.0f;
    if (a <= 1.0f) r += t1;
    if (a > 1.0f && a <= 2.0f) r += t2;
    return r;
}

__global__ void basis_kernel(const float* __restrict__ Cx,
                             const float* __restrict__ Cy,
                             const float* __restrict__ th0,
                             float* __restrict__ basisOut) {
    int idx = blockIdx.x * 256 + threadIdx.x;
    if (idx >= BATCH * SZP * SZP * TRAN * KPIX) return;
    int k = idx % KPIX;  int t = idx / KPIX;
    int o = t & 7;       t >>= 3;
    int j = t % SZP;     t /= SZP;
    int i = t % SZP;     int b = t / SZP;

    float xs_i = (i - 2) * 0.5f;           // xs = (-2..2)/2
    float xs_j = (j - 2) * 0.5f;
    float inX = xs_j, inY = xs_i;          // meshgrid 'xy'
    int   ai = k / 5, li = k % 5;
    float kv = (float)ai - 2.0f;
    float lv = (float)li - 2.0f;

    float cx = 1.0f + 0.01f * Cx[b];
    float cy = 1.0f + 0.01f * Cy[b];
    float t0 = 0.01f * th0[b];
    float c0 = cosf(t0), s0 = sinf(t0);

    float X  = c0 * inX - s0 * inY;
    float Y  = c0 * inY + s0 * inX;
    float X1 = X * cx, Y1 = Y * cy;
    float th = (float)o * 0.7853981633974483f;   // 2*pi/8
    float ct = cosf(th), st = sinf(th);
    float Xr = (ct * X1 - st * Y1) * 2.0f;        // * half(=INP//2)
    float Yr = (ct * Y1 + st * X1) * 2.0f;
    float bas = bicubic_f(Xr - kv) * bicubic_f(Yr - lv);

    // mask ('ij' meshgrid; bug-faithful: Yrm uses updated Xrm)
    float Xm = xs_i, Ym = xs_j;
    float Xrm = c0 * Xm - s0 * Ym;
    float Yrm = c0 * Ym + s0 * Xrm;
    float Cq  = (Xrm * cx) * (Xrm * cx) + (Yrm * cy) * (Yrm * cy);
    float mask = expf(-fmaxf(Cq - 1.0f, 0.0f) * 5.0f);   // /0.2

    basisOut[idx] = bas * mask;
}

// ---------------------------------------------------------------------------
// Stage 2: einsum + cyclic roll, f16 filters in layout [b][ky*5+kx][co][ci]
//   co = m*8 + o ; ci = n*8 + e ; pre-roll expand a = (e - o) mod 8
// ---------------------------------------------------------------------------
__global__ void filt_kernel(const float* __restrict__ basis,
                            const float* __restrict__ weights,
                            _Float16* __restrict__ filt) {
    int idx = blockIdx.x * 256 + threadIdx.x;   // 8*25*64*64 = 819200 exactly
    int ci = idx & 63;  int t = idx >> 6;
    int co = t & 63;    t >>= 6;
    int ij = t % KPIX;  int b = t / KPIX;
    int m = co >> 3, o = co & 7;
    int n = ci >> 3, e = ci & 7;
    int a = (e - o) & 7;
    const float* bp = basis + (size_t)((((b * SZP + ij / 5) * SZP + ij % 5) * TRAN + o) * KPIX);
    const float* wp = weights + (size_t)(((m * 8 + n) * 8 + a) * KPIX);
    float s = 0.0f;
#pragma unroll
    for (int kk = 0; kk < KPIX; ++kk) s += bp[kk] * wp[kk];
    filt[idx] = (_Float16)s;
}

// ---------------------------------------------------------------------------
// Stage 3: grouped conv as implicit GEMM with v_wmma_f32_16x16x32_f16.
// Block = (batch, row y, 64-pixel x tile). 8 waves: mt = wave>>1 (16 co),
// two 16-pixel n-tiles per wave. 25 kernel positions x 2 ci-chunks of 32.
// ---------------------------------------------------------------------------
__global__ void __launch_bounds__(256) conv_kernel(
        const float*    __restrict__ x,
        const _Float16* __restrict__ filt,
        const float*    __restrict__ cbias,
        float*          __restrict__ out) {
    __shared__ _Float16 lds[SZP * LDSCOLS * LDSPITCH];   // ~49 KB

    int blk = blockIdx.x;
    int xt  = blk % 3;   int t = blk / 3;
    int y   = t % HH;    int bb = t / HH;
    int x0  = xt * 64;
    int tid = threadIdx.x;

    // -- stage input patch (5 rows x 68 cols x 64 ci) as f16, ci contiguous --
    const float* xb = x + (size_t)bb * CIN * HH * WW;
    for (int it = 0; it < 85; ++it) {              // 5*64*68 = 21760 = 85*256
        int e   = tid + it * 256;                  // order [row][ci][col]
        int col = e % LDSCOLS;  int r2 = e / LDSCOLS;
        int ci  = r2 & 63;      int row = r2 >> 6;
        int gy  = y + row - 2;
        int gx  = x0 + col - 2;
        float v = 0.0f;
        if ((unsigned)gy < (unsigned)HH && (unsigned)gx < (unsigned)WW)
            v = xb[((size_t)ci * HH + gy) * WW + gx];
        lds[(row * LDSCOLS + col) * LDSPITCH + ci] = (_Float16)v;
    }
    __syncthreads();

    int lane = tid & 31;
    int wave = tid >> 5;
    int mt   = wave >> 1;          // 0..3  : 16-channel M tile
    int ntb  = (wave & 1) * 2;     // 0 or 2: first of two 16-pixel N tiles
    int hi   = lane >> 4;          // half-wave id (K striping)
    int lm   = lane & 15;

    v8f acc0 = {};
    v8f acc1 = {};

    // A operand base: layout [b][kykx][co][ci]; lane row co = mt*16+lm,
    // halves: run0 ci = c2*32 + hi*8 + 0..7, run1 = +16 (CDNA5 16-bit A map)
    const _Float16* wbase = filt + (size_t)bb * KPIX * COUT * CIN
                                 + (size_t)(mt * 16 + lm) * CIN + hi * 8;
    int colb0 = ntb * 16 + lm;     // LDS column of this lane's pixel (+kx)
    int colb1 = colb0 + 16;

    for (int ky = 0; ky < SZP; ++ky) {
        for (int kx = 0; kx < SZP; ++kx) {
            const _Float16* wk = wbase + (size_t)(ky * 5 + kx) * (COUT * CIN);
#pragma unroll
            for (int c2 = 0; c2 < 2; ++c2) {
                v8h alo = *(const v8h*)(wk + c2 * 32);
                v8h ahi = *(const v8h*)(wk + c2 * 32 + 16);
                v16h A = __builtin_shufflevector(alo, ahi,
                         0,1,2,3,4,5,6,7,8,9,10,11,12,13,14,15);

                const _Float16* bp0 =
                    &lds[(ky * LDSCOLS + colb0 + kx) * LDSPITCH + c2 * 32 + hi * 8];
                const _Float16* bp1 =
                    &lds[(ky * LDSCOLS + colb1 + kx) * LDSPITCH + c2 * 32 + hi * 8];
                v8h b0l = *(const v8h*)(bp0);
                v8h b0h = *(const v8h*)(bp0 + 16);
                v16h B0 = __builtin_shufflevector(b0l, b0h,
                          0,1,2,3,4,5,6,7,8,9,10,11,12,13,14,15);
                v8h b1l = *(const v8h*)(bp1);
                v8h b1h = *(const v8h*)(bp1 + 16);
                v16h B1 = __builtin_shufflevector(b1l, b1h,
                          0,1,2,3,4,5,6,7,8,9,10,11,12,13,14,15);

                acc0 = __builtin_amdgcn_wmma_f32_16x16x32_f16(
                        false, A, false, B0, (short)0, acc0, false, false);
                acc1 = __builtin_amdgcn_wmma_f32_16x16x32_f16(
                        false, A, false, B1, (short)0, acc1, false, false);
            }
        }
    }

    // D layout: VGPR r -> M = r + 8*hi, lane lm = pixel. One bias per lane.
    float bias = cbias[mt * 2 + hi];
    float* ob  = out + (((size_t)bb * COUT) * HH + y) * WW;
    int xp0    = x0 + ntb * 16 + lm;
#pragma unroll
    for (int r = 0; r < 8; ++r) {
        int co = mt * 16 + hi * 8 + r;
        ob[(size_t)co * HH * WW + xp0]      = acc0[r] + bias;
        ob[(size_t)co * HH * WW + xp0 + 16] = acc1[r] + bias;
    }
}

// ---------------------------------------------------------------------------
extern "C" void kernel_launch(void* const* d_in, const int* in_sizes, int n_in,
                              void* d_out, int out_size, void* d_ws, size_t ws_size,
                              hipStream_t stream) {
    const float* x   = (const float*)d_in[0];
    const float* Cx  = (const float*)d_in[1];
    const float* Cy  = (const float*)d_in[2];
    const float* th0 = (const float*)d_in[3];
    const float* w   = (const float*)d_in[4];
    const float* c   = (const float*)d_in[5];
    float* out = (float*)d_out;

    float*    basis = (float*)d_ws;                               // 160000 B
    _Float16* filt  = (_Float16*)((char*)d_ws + 160000);          // 1638400 B

    basis_kernel<<<157, 256, 0, stream>>>(Cx, Cy, th0, basis);    // 40000 elems
    filt_kernel<<<3200, 256, 0, stream>>>(basis, w, filt);        // 819200 elems
    conv_kernel<<<BATCH * HH * 3, 256, 0, stream>>>(x, filt, c, out);
}
